// AttentionStructureModel_26886495273181
// MI455X (gfx1250) — matrix operand order repaired and checked by reading, not compiled, not run on hardware
//
#include <hip/hip_runtime.h>
#include <hip/hip_bf16.h>
#include <math.h>

// ---------------------------------------------------------------------------
// CDNA5 (gfx1250) implementation. wave32. Matrix math via
// V_WMMA_F32_16X16X32_F16 (f16 in, f32 accumulate).
// ---------------------------------------------------------------------------

typedef __attribute__((ext_vector_type(16))) _Float16 v16h;
typedef __attribute__((ext_vector_type(8)))  float    v8f;

#define NB 384   // batch (sequences)
#define LL 256   // sequence length
#define HH 64    // hidden

__device__ __forceinline__ v8f wmma32(v16h a, v16h b, v8f c) {
  // 8 args: (neg_a, A, neg_b, B, c_mod, C, reuse_a, reuse_b)
  return __builtin_amdgcn_wmma_f32_16x16x32_f16(false, a, false, b, (short)0, c,
                                                false, false);
}

// A fragment: 16x32 f16 tile, row-major source with leading dim lda.
// Layout: lane L holds row M=L&15; lanes<16 carry K 0..15,
// lanes>=16 carry K 16..31; VGPR r holds one K pair.
__device__ __forceinline__ v16h load_a_frag(const _Float16* __restrict__ src,
                                            int lda, int lane) {
  v16h a;
  int m  = lane & 15;
  int kh = (lane >> 4) << 4;
  const _Float16* p = src + (size_t)m * lda + kh;
#pragma unroll
  for (int r = 0; r < 8; ++r) {
    int kb = r << 1;
    a[2 * r]     = p[kb];
    a[2 * r + 1] = p[kb + 1];
  }
  return a;
}

// B fragment (32x16) where B[k][n] = W[n][k] (i.e. computing X @ W^T).
// w points at W row (colTileBase), ldw = K stride of W. lanes<16: K 0..15,
// lanes>=16: K 16..31; N = lane&15. 16 contiguous halves per lane.
__device__ __forceinline__ v16h load_b_wt(const _Float16* __restrict__ w,
                                          int ldw, int lane) {
  int n  = lane & 15;
  int kb = (lane >> 4) << 4;
  const _Float16* p = w + (size_t)n * ldw + kb;
  v16h b;
#pragma unroll
  for (int i = 0; i < 16; ++i) b[i] = p[i];
  return b;
}

// B fragment from a genuinely row-major KxN source (leading dim ldn).
__device__ __forceinline__ v16h load_b_rm(const _Float16* __restrict__ src,
                                          int ldn, int lane) {
  int n  = lane & 15;
  int kb = (lane >> 4) << 4;
  const _Float16* p = src + (size_t)kb * ldn + n;
  v16h b;
#pragma unroll
  for (int i = 0; i < 16; ++i) b[i] = p[(size_t)i * ldn];
  return b;
}

// ---------------------------------------------------------------------------
// small helpers
// ---------------------------------------------------------------------------
__global__ void cvt_f16_kernel(const float* __restrict__ s,
                               _Float16* __restrict__ d, int n) {
  int i = blockIdx.x * blockDim.x + threadIdx.x;
  if (i < n) d[i] = (_Float16)s[i];
}

__global__ void pad_wih0_kernel(const float* __restrict__ s,
                                _Float16* __restrict__ d) {
  int i = blockIdx.x * blockDim.x + threadIdx.x;  // 192*32
  if (i < 192 * 32) {
    int r = i >> 5, c = i & 31;
    d[i] = (c < 30) ? (_Float16)s[r * 30 + c] : (_Float16)0.f;
  }
}

// ---------------------------------------------------------------------------
// featurize: RBF / Chebyshev / embedding select -> feat[N*L][32] f32 (pad 2)
// ---------------------------------------------------------------------------
__global__ void featurize_kernel(const float* __restrict__ x,
                                 const float* __restrict__ emb,
                                 float* __restrict__ feat) {
  int idx = blockIdx.x * blockDim.x + threadIdx.x;
  if (idx >= NB * LL) return;
  int n = idx / LL, l = idx % LL;
  const float* xb = x + (size_t)n * 3 * LL;
  float a0 = xb[l], a1 = xb[LL + l], a2 = xb[2 * LL + l];
  float* f = feat + (size_t)idx * 32;
  if ((n % 3) == 2) {  // angle row
#pragma unroll
    for (int i = 0; i < 10; ++i) { float d = (float)(i + 1) - a0; f[i] = expf(-d * d); }
#pragma unroll
    for (int i = 0; i < 10; ++i) { float d = (float)(i + 1) - a1; f[10 + i] = expf(-d * d); }
    float t0 = 1.f, t1 = a2;
    f[20] = t0; f[21] = t1;
#pragma unroll
    for (int i = 2; i < 10; ++i) { float t2 = 2.f * a2 * t1 - t0; f[20 + i] = t2; t0 = t1; t1 = t2; }
  } else {             // bond row
    int i1 = min(max((int)a0, 0), 118);
    int i2 = min(max((int)a1, 0), 118);
#pragma unroll
    for (int i = 0; i < 10; ++i) f[i] = emb[i1 * 10 + i];
#pragma unroll
    for (int i = 0; i < 10; ++i) f[10 + i] = emb[i2 * 10 + i];
#pragma unroll
    for (int i = 0; i < 10; ++i) { float d = (float)(i + 1) - a2; f[20 + i] = expf(-d * d); }
  }
  f[30] = 0.f; f[31] = 0.f;
}

// ---------------------------------------------------------------------------
// Generic WMMA GEMM: C[M][N] = act(A_f32[M][K] @ W_f16[N][K]^T + bias)
// grid (M/64, N/64); block 256 = 8 waves; each wave owns 2 adjacent N-tiles.
// A tile staged in LDS with contiguous 8-elem/thread chunks (b128 vectorizable)
// ---------------------------------------------------------------------------
__global__ void gemm_wmma_kernel(const float* __restrict__ A,
                                 const _Float16* __restrict__ W,
                                 const float* __restrict__ bias,
                                 float* __restrict__ C,
                                 int M, int N, int K, int act) {
  extern __shared__ _Float16 smA[];  // 64 x 32 f16
  const int lane = threadIdx.x & 31;
  const int wave = threadIdx.x >> 5;
  const int mi = wave >> 1;           // 0..3
  const int nb = (wave & 1) * 2;      // 0 or 2 (tile index within 4)
  const int row0 = blockIdx.x * 64;
  const int col0 = blockIdx.y * 64;
  // each thread stages 8 contiguous elements: row = tid>>2, cols (tid&3)*8..+7
  const float* asrc0 = A + (size_t)(row0 + (threadIdx.x >> 2)) * K +
                       ((threadIdx.x & 3) << 3);
  _Float16* adst = smA + threadIdx.x * 8;
  v8f acc0 = {}, acc1 = {};
  for (int k0 = 0; k0 < K; k0 += 32) {
    {
      const float* src = asrc0 + k0;
#pragma unroll
      for (int i = 0; i < 8; ++i) adst[i] = (_Float16)src[i];
      if (k0 + 32 < K) __builtin_prefetch(src + 32, 0, 1);
    }
    __syncthreads();
    v16h a  = load_a_frag(smA + mi * 16 * 32, 32, lane);
    v16h b0 = load_b_wt(W + (size_t)(col0 + nb * 16) * K + k0, K, lane);
    v16h b1 = load_b_wt(W + (size_t)(col0 + (nb + 1) * 16) * K + k0, K, lane);
    acc0 = wmma32(a, b0, acc0);
    acc1 = wmma32(a, b1, acc1);
    __syncthreads();
  }
  const int rsub = (lane >> 4) << 3;
  const int csub = lane & 15;
#pragma unroll
  for (int r = 0; r < 8; ++r) {
    int gr = row0 + mi * 16 + r + rsub;
    int gc0 = col0 + nb * 16 + csub;
    int gc1 = col0 + (nb + 1) * 16 + csub;
    float v0 = acc0[r] + bias[gc0];
    float v1 = acc1[r] + bias[gc1];
    if (act == 1) { v0 = fmaxf(v0, 0.f); v1 = fmaxf(v1, 0.f); }
    C[(size_t)gr * N + gc0] = v0;
    C[(size_t)gr * N + gc1] = v1;
  }
}

// ---------------------------------------------------------------------------
// GRU scan: h_t = GRU(h_{t-1}, gi[t]) over L steps. 16 batch rows / block,
// 128 threads (4 waves). gh = h @ whh^T: 24 WMMAs per step, whh cached in LDS.
// gi[t] preloaded into registers BEFORE the matmul so the global latency
// overlaps the WMMA chain; h + bhh live in registers across all 256 steps.
// ---------------------------------------------------------------------------
__global__ void gru_scan_kernel(const float* __restrict__ gi,      // [N][L][192]
                                const _Float16* __restrict__ whh,  // [192][64]
                                const float* __restrict__ bhh,     // [192]
                                float* __restrict__ out) {         // [N][L][64]
  __shared__ __align__(16) _Float16 wsh[192 * 64];  // 24 KB
  __shared__ __align__(16) _Float16 h16[16 * 64];   // 2 KB
  __shared__ float gh[16 * 192];                    // 12 KB
  const int tid = threadIdx.x;
  const int lane = tid & 31, wave = tid >> 5;
  const int b0 = blockIdx.x * 16;
  // stage whh into LDS (int4 = 8 halves per copy)
  {
    const int4* s = (const int4*)whh;
    int4* d = (int4*)wsh;
    for (int i = tid; i < (192 * 64) / 8; i += 128) d[i] = s[i];
  }
  {
    int4 z = {0, 0, 0, 0};
    ((int4*)h16)[tid] = z;  // 128 * 8 halves = full clear
  }
  // static per-thread gate ownership: row = tid>>3, cols c0..c0+7
  const int row = tid >> 3;
  const int c0 = (tid & 7) << 3;
  float bhr[8], bhz[8], bhn[8], hreg[8];
#pragma unroll
  for (int i = 0; i < 8; ++i) {
    bhr[i] = bhh[c0 + i];
    bhz[i] = bhh[64 + c0 + i];
    bhn[i] = bhh[128 + c0 + i];
    hreg[i] = 0.f;
  }
  const float* gi_row = gi + ((size_t)(b0 + row) * LL) * 192 + c0;
  float* out_row = out + ((size_t)(b0 + row) * LL) * 64 + c0;
  __syncthreads();
  for (int t = 0; t < LL; ++t) {
    // issue gate-input loads early (consumed after the barrier)
    const float* gp = gi_row + (size_t)t * 192;
    float gir[8], giz[8], gin[8];
#pragma unroll
    for (int i = 0; i < 8; ++i) {
      gir[i] = gp[i];
      giz[i] = gp[64 + i];
      gin[i] = gp[128 + i];
    }
    if (t + 1 < LL) {
      __builtin_prefetch(gp + 192, 0, 1);
      __builtin_prefetch(gp + 192 + 64, 0, 1);
      __builtin_prefetch(gp + 192 + 128, 0, 1);
    }
    // gh = h @ whh^T
    v8f acc[3] = {{}, {}, {}};
#pragma unroll
    for (int ks = 0; ks < 2; ++ks) {
      v16h a = load_a_frag(h16 + ks * 32, 64, lane);
#pragma unroll
      for (int j = 0; j < 3; ++j) {
        int ni = wave * 3 + j;  // 12 output tiles of 16 cols
        v16h bb = load_b_wt(wsh + (size_t)(ni * 16) * 64 + ks * 32, 64, lane);
        acc[j] = wmma32(a, bb, acc[j]);
      }
    }
    const int rsub = (lane >> 4) << 3;
    const int csub = lane & 15;
#pragma unroll
    for (int j = 0; j < 3; ++j) {
      int ni = wave * 3 + j;
#pragma unroll
      for (int r = 0; r < 8; ++r)
        gh[(r + rsub) * 192 + ni * 16 + csub] = acc[j][r];
    }
    __syncthreads();
    // gates (contiguous 8-wide chunks -> vectorized LDS/global traffic)
    float* orow = out_row + (size_t)t * 64;
    _Float16* hrow = h16 + row * 64 + c0;
    const float* ghrow = gh + row * 192 + c0;
#pragma unroll
    for (int i = 0; i < 8; ++i) {
      float ghr = ghrow[i]       + bhr[i];
      float ghz = ghrow[64 + i]  + bhz[i];
      float ghn = ghrow[128 + i] + bhn[i];
      float rr = 1.f / (1.f + expf(-(gir[i] + ghr)));
      float zz = 1.f / (1.f + expf(-(giz[i] + ghz)));
      float nn = tanhf(gin[i] + rr * ghn);
      float hn = (1.f - zz) * nn + zz * hreg[i];
      hreg[i] = hn;
      hrow[i] = (_Float16)hn;
      orow[i] = hn;
    }
    __syncthreads();
  }
}

// ---------------------------------------------------------------------------
// Attention core per (batch, head): scores (WMMA) -> softmax in LDS -> P@V
// block 256 (8 waves). LDS: q/k/v f16 (48KB) + full 256x256 f16 scores (128KB)
// ---------------------------------------------------------------------------
__global__ void attn_kernel(const float* __restrict__ qkv,   // [N][L][192]
                            float* __restrict__ oheads) {    // [N][L][64]
  extern __shared__ _Float16 smem[];
  _Float16* qs = smem;                 // 256*32
  _Float16* ks = qs + 256 * 32;
  _Float16* vs = ks + 256 * 32;
  _Float16* sc = vs + 256 * 32;        // 256*256
  float* rinv = (float*)(sc + 256 * 256);  // 256
  const int b = blockIdx.x >> 1, hidx = blockIdx.x & 1;
  const int tid = threadIdx.x, lane = tid & 31, wave = tid >> 5;
  const float scale = 0.17677669529663687f;  // 1/sqrt(32)
  {
    // one full row (32 contiguous values per array) per thread
    const float* src = qkv + ((size_t)b * LL + tid) * 192 + hidx * 32;
    _Float16* q = qs + tid * 32;
    _Float16* kk = ks + tid * 32;
    _Float16* vv = vs + tid * 32;
#pragma unroll
    for (int c = 0; c < 32; ++c) {
      q[c]  = (_Float16)(src[c] * scale);
      kk[c] = (_Float16)(src[64 + c]);
      vv[c] = (_Float16)(src[128 + c]);
    }
  }
  __syncthreads();
  // scores: 16x16 row tiles x 16 col tiles; K = 32 (one WMMA per tile)
  const int rsub = (lane >> 4) << 3;
  const int csub = lane & 15;
  for (int mi = wave * 2; mi < wave * 2 + 2; ++mi) {
    v16h a = load_a_frag(qs + mi * 16 * 32, 32, lane);
    for (int ni = 0; ni < 16; ++ni) {
      v16h bb = load_b_wt(ks + ni * 16 * 32, 32, lane);  // k^T
      v8f accs = {};
      accs = wmma32(a, bb, accs);
#pragma unroll
      for (int r = 0; r < 8; ++r)
        sc[(size_t)(mi * 16 + r + rsub) * 256 + ni * 16 + csub] = (_Float16)accs[r];
    }
  }
  __syncthreads();
  // softmax: one thread per row (unnormalized exp in LDS, 1/sum kept aside)
  {
    _Float16* srow = sc + (size_t)tid * 256;
    float m = -1e30f;
    for (int j = 0; j < 256; ++j) m = fmaxf(m, (float)srow[j]);
    float s = 0.f;
    for (int j = 0; j < 256; ++j) {
      float p = expf((float)srow[j] - m);
      srow[j] = (_Float16)p;
      s += p;
    }
    rinv[tid] = 1.f / s;
  }
  __syncthreads();
  // o = P @ V : 16 row tiles x 2 col tiles, K=256
  for (int tI = wave * 4; tI < wave * 4 + 4; ++tI) {
    int mi = tI >> 1, ni = tI & 1;
    v8f acc = {};
    for (int k0 = 0; k0 < 256; k0 += 32) {
      v16h a  = load_a_frag(sc + (size_t)mi * 16 * 256 + k0, 256, lane);
      v16h bb = load_b_rm(vs + (size_t)k0 * 32 + ni * 16, 32, lane);
      acc = wmma32(a, bb, acc);
    }
#pragma unroll
    for (int r = 0; r < 8; ++r) {
      int row = mi * 16 + r + rsub;
      float v = acc[r] * rinv[row];
      oheads[((size_t)b * LL + row) * 64 + hidx * 32 + ni * 16 + csub] = v;
    }
  }
}

// ---------------------------------------------------------------------------
// Fused FFN: out = relu(relu(X@W1^T + b1)@W2^T + b2), hidden tile in LDS
// block 256 (8 waves), 64 rows per block.
// ---------------------------------------------------------------------------
__global__ void ffn_kernel(const float* __restrict__ X,
                           const _Float16* __restrict__ W1,  // [1024][64]
                           const float* __restrict__ b1,
                           const _Float16* __restrict__ W2,  // [64][1024]
                           const float* __restrict__ b2,
                           float* __restrict__ out) {
  extern __shared__ _Float16 smem[];
  _Float16* xs  = smem;           // 64*64
  _Float16* hid = xs + 64 * 64;   // 64*1024
  const int tid = threadIdx.x, lane = tid & 31, wave = tid >> 5;
  const int row0 = blockIdx.x * 64;
  {
    // 16 contiguous elements per thread (row = tid>>2, cols (tid&3)*16..+15)
    const float* src = X + (size_t)row0 * 64 + tid * 16;
    _Float16* dst = xs + tid * 16;
#pragma unroll
    for (int i = 0; i < 16; ++i) dst[i] = (_Float16)src[i];
  }
  __syncthreads();
  const int rsub = (lane >> 4) << 3;
  const int csub = lane & 15;
  {  // hidden: 4 row-tiles x 64 col-tiles, K=64
    int mi = wave >> 1;
    v16h a0 = load_a_frag(xs + mi * 16 * 64, 64, lane);
    v16h a1 = load_a_frag(xs + mi * 16 * 64 + 32, 64, lane);
    for (int ni = (wave & 1) * 32; ni < (wave & 1) * 32 + 32; ++ni) {
      v16h bb0 = load_b_wt(W1 + (size_t)(ni * 16) * 64, 64, lane);
      v16h bb1 = load_b_wt(W1 + (size_t)(ni * 16) * 64 + 32, 64, lane);
      v8f acc = {};
      acc = wmma32(a0, bb0, acc);
      acc = wmma32(a1, bb1, acc);
#pragma unroll
      for (int r = 0; r < 8; ++r) {
        float v = acc[r] + b1[ni * 16 + csub];
        v = fmaxf(v, 0.f);
        hid[(size_t)(mi * 16 + r + rsub) * 1024 + ni * 16 + csub] = (_Float16)v;
      }
    }
  }
  __syncthreads();
  // out: 4x4 tiles, K=1024
  for (int tI = wave * 2; tI < wave * 2 + 2; ++tI) {
    int mi = tI >> 2, ni = tI & 3;
    v8f acc = {};
    for (int k0 = 0; k0 < 1024; k0 += 32) {
      v16h a  = load_a_frag(hid + (size_t)mi * 16 * 1024 + k0, 1024, lane);
      v16h bb = load_b_wt(W2 + (size_t)(ni * 16) * 1024 + k0, 1024, lane);
      acc = wmma32(a, bb, acc);
    }
#pragma unroll
    for (int r = 0; r < 8; ++r) {
      float v = acc[r] + b2[ni * 16 + csub];
      v = fmaxf(v, 0.f);
      out[(size_t)(row0 + mi * 16 + r + rsub) * 64 + ni * 16 + csub] = v;
    }
  }
}

// ---------------------------------------------------------------------------
// LayerNorm over H=64 of (A + B): one wave per row, 2 elements / lane
// ---------------------------------------------------------------------------
__global__ void ln_kernel(const float* __restrict__ A, const float* __restrict__ B,
                          const float* __restrict__ s, const float* __restrict__ bb,
                          float* __restrict__ out, int M) {
  int wave = threadIdx.x >> 5, lane = threadIdx.x & 31;
  int row = blockIdx.x * 8 + wave;
  if (row >= M) return;
  const float* a = A + (size_t)row * 64;
  const float* b = B + (size_t)row * 64;
  float x0 = a[lane] + b[lane];
  float x1 = a[lane + 32] + b[lane + 32];
  float sum = x0 + x1;
#pragma unroll
  for (int off = 16; off > 0; off >>= 1) sum += __shfl_xor(sum, off, 32);
  float mean = sum * (1.f / 64.f);
  float d0 = x0 - mean, d1 = x1 - mean;
  float vs = d0 * d0 + d1 * d1;
#pragma unroll
  for (int off = 16; off > 0; off >>= 1) vs += __shfl_xor(vs, off, 32);
  float inv = rsqrtf(vs * (1.f / 64.f) + 1e-5f);
  out[(size_t)row * 64 + lane]      = d0 * inv * s[lane] + bb[lane];
  out[(size_t)row * 64 + lane + 32] = d1 * inv * s[lane + 32] + bb[lane + 32];
}

// ---------------------------------------------------------------------------
// Head MLP on h[:, L-1, :] : 64 -> 256 -> 64 -> 32 -> 1 with SiLU
// ---------------------------------------------------------------------------
__global__ void head_kernel(const float* __restrict__ H,
                            const float* __restrict__ fw1, const float* __restrict__ fb1,
                            const float* __restrict__ fw2, const float* __restrict__ fb2,
                            const float* __restrict__ fw3, const float* __restrict__ fb3,
                            const float* __restrict__ fw4, const float* __restrict__ fb4,
                            float* __restrict__ out) {
  __shared__ float in[64], o1[256], o2[64], o3[32];
  int b = blockIdx.x, tid = threadIdx.x;  // block 64
  in[tid] = H[((size_t)b * LL + (LL - 1)) * 64 + tid];
  __syncthreads();
  for (int j = tid; j < 256; j += 64) {
    float acc = fb1[j];
    for (int k = 0; k < 64; ++k) acc += in[k] * fw1[j * 64 + k];
    o1[j] = acc / (1.f + expf(-acc));
  }
  __syncthreads();
  {
    float acc = fb2[tid];
    for (int k = 0; k < 256; ++k) acc += o1[k] * fw2[tid * 256 + k];
    o2[tid] = acc / (1.f + expf(-acc));
  }
  __syncthreads();
  if (tid < 32) {
    float acc = fb3[tid];
    for (int k = 0; k < 64; ++k) acc += o2[k] * fw3[tid * 64 + k];
    o3[tid] = acc / (1.f + expf(-acc));
  }
  __syncthreads();
  if (tid == 0) {
    float acc = fb4[0];
    for (int k = 0; k < 32; ++k) acc += o3[k] * fw4[k];
    out[b] = acc;
  }
}

// ---------------------------------------------------------------------------
// Launch orchestration
// ---------------------------------------------------------------------------
extern "C" void kernel_launch(void* const* d_in, const int* in_sizes, int n_in,
                              void* d_out, int out_size, void* d_ws, size_t ws_size,
                              hipStream_t stream) {
  (void)in_sizes; (void)n_in; (void)out_size; (void)ws_size;
  const float* x      = (const float*)d_in[0];
  const float* emb    = (const float*)d_in[1];
  const float* wih0   = (const float*)d_in[2];
  const float* whh0   = (const float*)d_in[3];
  const float* bih0   = (const float*)d_in[4];
  const float* bhh0   = (const float*)d_in[5];
  const float* wih12  = (const float*)d_in[6];
  const float* whh12  = (const float*)d_in[7];
  const float* bih12  = (const float*)d_in[8];
  const float* bhh12  = (const float*)d_in[9];
  const float* in_w1  = (const float*)d_in[10];
  const float* in_b1  = (const float*)d_in[11];
  const float* out_w1 = (const float*)d_in[12];
  const float* out_b1 = (const float*)d_in[13];
  const float* in_w2  = (const float*)d_in[14];
  const float* in_b2  = (const float*)d_in[15];
  const float* out_w2 = (const float*)d_in[16];
  const float* out_b2 = (const float*)d_in[17];
  const float* ff_w1  = (const float*)d_in[18];
  const float* ff_b1  = (const float*)d_in[19];
  const float* ff_w2  = (const float*)d_in[20];
  const float* ff_b2  = (const float*)d_in[21];
  const float* ln1_s  = (const float*)d_in[22];
  const float* ln1_b  = (const float*)d_in[23];
  const float* ln2_s  = (const float*)d_in[24];
  const float* ln2_b  = (const float*)d_in[25];
  const float* ln3_s  = (const float*)d_in[26];
  const float* ln3_b  = (const float*)d_in[27];
  const float* ln4_s  = (const float*)d_in[28];
  const float* ln4_b  = (const float*)d_in[29];
  const float* fw1    = (const float*)d_in[30];
  const float* fb1    = (const float*)d_in[31];
  const float* fw2    = (const float*)d_in[32];
  const float* fb2    = (const float*)d_in[33];
  const float* fw3    = (const float*)d_in[34];
  const float* fb3    = (const float*)d_in[35];
  const float* fw4    = (const float*)d_in[36];
  const float* fb4    = (const float*)d_in[37];
  float* outp = (float*)d_out;

  const size_t M = (size_t)NB * LL;  // 98304 rows

  // ---- workspace carve-out ----
  char* ws = (char*)d_ws;
  size_t off = 0;
  auto carve = [&](size_t bytes) -> void* {
    void* p = ws + off;
    off += (bytes + 255) & ~(size_t)255;
    return p;
  };
  float* feat = (float*)carve(M * 32 * 4);    // 12.6 MB
  float* gi   = (float*)carve(M * 192 * 4);   // 75.5 MB (also qkv)
  float* seqA = (float*)carve(M * 64 * 4);    // 25.2 MB
  float* seqB = (float*)carve(M * 64 * 4);
  float* seqC = (float*)carve(M * 64 * 4);
  _Float16* w16pool = (_Float16*)carve(262144 * 2);
  _Float16* wih0_16  = w16pool;                        // 192*32
  _Float16* whh0_16  = wih0_16 + 192 * 32;             // 192*64
  _Float16* wih12_16 = whh0_16 + 192 * 64;             // 2*192*64
  _Float16* whh12_16 = wih12_16 + 2 * 192 * 64;        // 2*192*64
  _Float16* in_w1_16 = whh12_16 + 2 * 192 * 64;        // 192*64
  _Float16* out_w1_16 = in_w1_16 + 192 * 64;           // 64*64
  _Float16* in_w2_16 = out_w1_16 + 64 * 64;            // 192*64
  _Float16* out_w2_16 = in_w2_16 + 192 * 64;           // 64*64
  _Float16* ff_w1_16 = out_w2_16 + 64 * 64;            // 1024*64
  _Float16* ff_w2_16 = ff_w1_16 + 1024 * 64;           // 64*1024

  auto cvt = [&](const float* s, _Float16* d, int n) {
    cvt_f16_kernel<<<(n + 255) / 256, 256, 0, stream>>>(s, d, n);
  };

  // ---- weight conversion (f32 -> f16) ----
  pad_wih0_kernel<<<(192 * 32 + 255) / 256, 256, 0, stream>>>(wih0, wih0_16);
  cvt(whh0, whh0_16, 192 * 64);
  cvt(wih12, wih12_16, 2 * 192 * 64);
  cvt(whh12, whh12_16, 2 * 192 * 64);
  cvt(in_w1, in_w1_16, 192 * 64);
  cvt(out_w1, out_w1_16, 64 * 64);
  cvt(in_w2, in_w2_16, 192 * 64);
  cvt(out_w2, out_w2_16, 64 * 64);
  cvt(ff_w1, ff_w1_16, 1024 * 64);
  cvt(ff_w2, ff_w2_16, 64 * 1024);

  // ---- featurize ----
  featurize_kernel<<<(NB * LL + 255) / 256, 256, 0, stream>>>(x, emb, feat);

  const dim3 gGemm192((unsigned)(M / 64), 3);
  const dim3 gGemm64((unsigned)(M / 64), 1);
  const size_t smGemm = 64 * 32 * sizeof(_Float16);
  const size_t smAttn = (3 * 256 * 32 + 256 * 256) * sizeof(_Float16) + 256 * sizeof(float);
  const size_t smFfn  = (64 * 64 + 64 * 1024) * sizeof(_Float16);

  // ---- GRU stack ----
  // layer 0: gi = feat @ wih0^T + bih0 ; scan -> seqA
  gemm_wmma_kernel<<<gGemm192, 256, smGemm, stream>>>(feat, wih0_16, bih0, gi,
                                                      (int)M, 192, 32, 0);
  gru_scan_kernel<<<NB / 16, 128, 0, stream>>>(gi, whh0_16, bhh0, seqA);
  // layer 1
  gemm_wmma_kernel<<<gGemm192, 256, smGemm, stream>>>(seqA, wih12_16, bih12, gi,
                                                      (int)M, 192, 64, 0);
  gru_scan_kernel<<<NB / 16, 128, 0, stream>>>(gi, whh12_16, bhh12, seqB);
  // layer 2
  gemm_wmma_kernel<<<gGemm192, 256, smGemm, stream>>>(seqB, wih12_16 + 192 * 64,
                                                      bih12 + 192, gi,
                                                      (int)M, 192, 64, 0);
  gru_scan_kernel<<<NB / 16, 128, 0, stream>>>(gi, whh12_16 + 192 * 64,
                                               bhh12 + 192, seqA);
  // seqA = g (GRU output, [N][L][64])

  // ---- transformer block 1 ----
  gemm_wmma_kernel<<<gGemm192, 256, smGemm, stream>>>(seqA, in_w1_16, in_b1, gi,
                                                      (int)M, 192, 64, 0);      // qkv
  attn_kernel<<<NB * 2, 256, smAttn, stream>>>(gi, seqB);                       // heads
  gemm_wmma_kernel<<<gGemm64, 256, smGemm, stream>>>(seqB, out_w1_16, out_b1,
                                                     seqC, (int)M, 64, 64, 0);  // a
  ln_kernel<<<(unsigned)((M + 7) / 8), 256, 0, stream>>>(seqA, seqC, ln1_s, ln1_b,
                                                         seqB, (int)M);         // h=LN(g+a)
  ffn_kernel<<<(unsigned)(M / 64), 256, smFfn, stream>>>(seqB, ff_w1_16, ff_b1,
                                                         ff_w2_16, ff_b2, seqC);
  ln_kernel<<<(unsigned)((M + 7) / 8), 256, 0, stream>>>(seqB, seqC, ln2_s, ln2_b,
                                                         seqA, (int)M);         // h=LN(h+ffn)

  // ---- transformer block 2 ----
  gemm_wmma_kernel<<<gGemm192, 256, smGemm, stream>>>(seqA, in_w2_16, in_b2, gi,
                                                      (int)M, 192, 64, 0);      // qkv
  attn_kernel<<<NB * 2, 256, smAttn, stream>>>(gi, seqB);
  gemm_wmma_kernel<<<gGemm64, 256, smGemm, stream>>>(seqB, out_w2_16, out_b2,
                                                     seqC, (int)M, 64, 64, 0);  // a
  // NOTE: reference does LN(a + a) here (residual dropped) — replicate exactly
  ln_kernel<<<(unsigned)((M + 7) / 8), 256, 0, stream>>>(seqC, seqC, ln3_s, ln3_b,
                                                         seqA, (int)M);
  ffn_kernel<<<(unsigned)(M / 64), 256, smFfn, stream>>>(seqA, ff_w1_16, ff_b1,
                                                         ff_w2_16, ff_b2, seqB);
  ln_kernel<<<(unsigned)((M + 7) / 8), 256, 0, stream>>>(seqA, seqB, ln4_s, ln4_b,
                                                         seqC, (int)M);

  // ---- head MLP on last timestep ----
  head_kernel<<<NB, 64, 0, stream>>>(seqC, fw1, fb1, fw2, fb2, fw3, fb3, fw4, fb4,
                                     outp);
}
